// LigerFusedLinearGRPOLoss_29841432773164
// MI455X (gfx1250) — matrix operand order repaired and checked by reading, not compiled, not run on hardware
//
#include <hip/hip_runtime.h>
#include <hip/hip_bf16.h>
#include <stdint.h>

// ---------------------------------------------------------------------------
// Fused Linear GRPO loss for MI455X (gfx1250, wave32, WMMA).
//
// Roofline: 1.07 TFLOP vs ~590 MB HBM -> strongly compute bound.
//   - GEMM in bf16 WMMA (v_wmma_f32_16x16x32_bf16, f32 accumulate).
//   - bf16 weights (131 MB) fit the 192 MB L2: convert once, stream from L2.
//   - Each wave owns a 2x2 grid of 16x16 tiles: 4 WMMAs per (A,B) frag pair
//     -> 2 x b128 loads per WMMA, halving L2/LDS bandwidth per MAC.
//   - Online softmax fused into the N sweep (no logits materialization);
//     argmax carried as a branchless 64-bit sort key.
// ---------------------------------------------------------------------------

typedef __attribute__((ext_vector_type(16))) __bf16 v16bf;
typedef __attribute__((ext_vector_type(8)))  __bf16 v8bf;
typedef __attribute__((ext_vector_type(8)))  float  v8f;

#define TOKENS  4096
#define HDIM    2048
#define VDIM    32000
#define SS      1024         // sequence length (tokens per batch row)
#define M_TILE  32           // token rows per workgroup (2 m-tiles)
#define N_STEP  256          // columns per iteration (8 waves x 2 n-tiles x 16)
#define NSPLIT  5            // V split across blockIdx.y (6400 = 25*256 each)
#define LDA_H   2056         // LDS A pitch in halves (2048+8 pad: rows start 4 banks apart)
#define SCR_ST  260          // softmax scratch pitch in floats (256+4 pad)

__device__ __forceinline__ uint16_t f2bf(float f) {
    uint32_t u = __float_as_uint(f);
    uint32_t r = u + 0x7FFFu + ((u >> 16) & 1u);   // round-to-nearest-even
    return (uint16_t)(r >> 16);
}
// monotonic float<->uint32 order map (for branchless argmax keys)
__device__ __forceinline__ uint32_t ford(float f) {
    uint32_t u = __float_as_uint(f);
    return u ^ (uint32_t)(((int32_t)u >> 31) | 0x80000000);
}
__device__ __forceinline__ float funord(uint32_t o) {
    uint32_t x = 0x80000000u | ~(uint32_t)((int32_t)o >> 31);
    return __uint_as_float(o ^ x);
}

// ---------------- weight f32 -> bf16 (one HBM sweep, L2-resident afterwards)
__global__ void cvt_f32_bf16(const float* __restrict__ src,
                             uint16_t* __restrict__ dst, long n) {
    long i = (long)blockIdx.x * blockDim.x + threadIdx.x;
    long stride = (long)gridDim.x * blockDim.x;
    const float2* s2 = (const float2*)src;
    uint32_t*     d2 = (uint32_t*)dst;
    long n2 = n >> 1;
    for (long j = i; j < n2; j += stride) {
        float2 v = s2[j];
        d2[j] = (uint32_t)f2bf(v.x) | ((uint32_t)f2bf(v.y) << 16);
    }
}

// ---------------- fused GEMM (WMMA bf16) + online softmax over an N slice
// MODE 0: policy -> partial (max, sumexp, argmax key) ; MODE 1: ref -> partial
// (max, sumexp, value at chosen index).
template <int MODE>
__global__ void fused_logits_softmax(
    const float*    __restrict__ x,       // (TOKENS, HDIM) f32 activations
    const uint16_t* __restrict__ wbf,     // (VDIM, HDIM)  bf16 weights
    const float*    __restrict__ bias,    // (VDIM)
    const int*      __restrict__ chosen,  // MODE 1: chosen index per token
    float* __restrict__ pmax, float* __restrict__ psum,
    float* __restrict__ pbv,  int*   __restrict__ pbi,   // MODE 0 partials
    float* __restrict__ pcap)                            // MODE 1 partials
{
    extern __shared__ char smem[];
    uint16_t* Abf = (uint16_t*)smem;                             // 32 x LDA_H halves
    float*    scr = (float*)(smem + (size_t)M_TILE * LDA_H * 2); // 16 x SCR_ST floats

    const int tid  = threadIdx.x;
    const int wave = tid >> 5;
    const int lane = tid & 31;
    const int kh   = lane >> 4;        // half-wave: selects K sub-chunk
    const int l15  = lane & 15;
    const int token0 = blockIdx.x * M_TILE;
    const int nbase  = blockIdx.y * (VDIM / NSPLIT);

    // ---- stage A: convert 32 rows of x to bf16 into LDS (full K = 2048)
    {
        const float2* x2  = (const float2*)x;
        uint32_t*     A32 = (uint32_t*)Abf;
        const int uPerRow = HDIM / 2;       // 1024
        const int ldsU    = LDA_H / 2;      // 1028
        for (int i = tid; i < M_TILE * uPerRow; i += blockDim.x) {
            int row = i >> 10;
            int cu  = i & 1023;
            float2 v = x2[(size_t)(token0 + row) * uPerRow + cu];
            A32[row * ldsU + cu] =
                (uint32_t)f2bf(v.x) | ((uint32_t)f2bf(v.y) << 16);
        }
    }
    __syncthreads();

    // A row bases for the two m-tiles this lane feeds
    const uint16_t* ab0 = Abf + (size_t)(l15)      * LDA_H;
    const uint16_t* ab1 = Abf + (size_t)(16 + l15) * LDA_H;

    // running per-row state; slot s -> row 16*(s>>1) + 2*wave + (s&1)
    float rmax[4], rsum[4], rcap[4];
    unsigned long long rkey[4];
    int rch[4];
#pragma unroll
    for (int s = 0; s < 4; ++s) {
        rmax[s] = -__builtin_inff(); rsum[s] = 0.f;
        rcap[s] = -__builtin_inff(); rkey[s] = 0ull;
        if (MODE == 1)
            rch[s] = chosen[token0 + 16 * (s >> 1) + 2 * wave + (s & 1)];
    }

    for (int n0 = nbase; n0 < nbase + VDIM / NSPLIT; n0 += N_STEP) {
        // this wave's two B columns (n-tiles 2*wave, 2*wave+1 of the 256 block)
        const int c0 = n0 + 32 * wave + l15;
        const uint16_t* wr0 = wbf + (size_t)c0 * HDIM;
        const uint16_t* wr1 = wr0 + (size_t)16 * HDIM;
        // pull next iteration's rows toward the caches (global_prefetch_b8)
        __builtin_prefetch(wr0 + (size_t)N_STEP * HDIM, 0, 3);
        __builtin_prefetch(wr1 + (size_t)N_STEP * HDIM, 0, 3);

        v8f acc[2][2];
#pragma unroll
        for (int mi = 0; mi < 2; ++mi)
#pragma unroll
            for (int nj = 0; nj < 2; ++nj)
                acc[mi][nj] = (v8f){0.f,0.f,0.f,0.f,0.f,0.f,0.f,0.f};

#pragma unroll 2
        for (int kb = 0; kb < HDIM; kb += 32) {
            // A 16x32 bf16: lanes 0-15 K{0..15}, lanes 16-31 K{16..31}
            v8bf a0l = *(const v8bf*)(ab0 + kb + kh * 16);
            v8bf a0h = *(const v8bf*)(ab0 + kb + 8 + kh * 16);
            v8bf a1l = *(const v8bf*)(ab1 + kb + kh * 16);
            v8bf a1h = *(const v8bf*)(ab1 + kb + 8 + kh * 16);
            v16bf a0 = __builtin_shufflevector(a0l, a0h,
                0,1,2,3,4,5,6,7,8,9,10,11,12,13,14,15);
            v16bf a1 = __builtin_shufflevector(a1l, a1h,
                0,1,2,3,4,5,6,7,8,9,10,11,12,13,14,15);
            // B 32x16 bf16: lane = N col, halves hold K 0..15 / 16..31 contiguous
            v16bf b0 = *(const v16bf*)(wr0 + kb + kh * 16);
            v16bf b1 = *(const v16bf*)(wr1 + kb + kh * 16);
            // 4 WMMAs per frag set -> 2 b128 loads per WMMA
            acc[0][0] = __builtin_amdgcn_wmma_f32_16x16x32_bf16(
                false, a0, false, b0, (short)0, acc[0][0], false, false);
            acc[0][1] = __builtin_amdgcn_wmma_f32_16x16x32_bf16(
                false, a0, false, b1, (short)0, acc[0][1], false, false);
            acc[1][0] = __builtin_amdgcn_wmma_f32_16x16x32_bf16(
                false, a1, false, b0, (short)0, acc[1][0], false, false);
            acc[1][1] = __builtin_amdgcn_wmma_f32_16x16x32_bf16(
                false, a1, false, b1, (short)0, acc[1][1], false, false);
        }

        const float bv0 = bias[c0];
        const float bv1 = bias[c0 + 16];

        // drain the 32x256 block tile in two 16-row phases through scratch
#pragma unroll
        for (int p = 0; p < 2; ++p) {
            // C layout: VGPR r -> local row r + 8*half, col = lane&15 of tile
#pragma unroll
            for (int r = 0; r < 8; ++r) {
                int lrow = r + 8 * kh;
                scr[lrow * SCR_ST + 32 * wave + l15]      = acc[p][0][r] + bv0;
                scr[lrow * SCR_ST + 32 * wave + 16 + l15] = acc[p][1][r] + bv1;
            }
            __syncthreads();

            // each wave reduces 2 rows of 256 values (8 per lane)
#pragma unroll
            for (int r = 0; r < 2; ++r) {
                const int slot = p * 2 + r;
                const int lrow = 2 * wave + r;
                float vv[8];
                if (MODE == 0) {
                    unsigned long long key = 0ull;
#pragma unroll
                    for (int c = 0; c < 8; ++c) {
                        vv[c] = scr[lrow * SCR_ST + lane + 32 * c];
                        uint32_t n = (uint32_t)(n0 + lane + 32 * c);
                        unsigned long long k =
                            ((unsigned long long)ford(vv[c]) << 32) | (uint32_t)~n;
                        key = k > key ? k : key;
                    }
#pragma unroll
                    for (int off = 16; off; off >>= 1) {
                        unsigned long long ok = __shfl_xor(key, off, 32);
                        key = ok > key ? ok : key;
                    }
                    float mv = funord((uint32_t)(key >> 32));
                    float s = 0.f;
#pragma unroll
                    for (int c = 0; c < 8; ++c) s += __expf(vv[c] - mv);
#pragma unroll
                    for (int off = 16; off; off >>= 1) s += __shfl_xor(s, off, 32);
                    float nm = fmaxf(rmax[slot], mv);
                    rsum[slot] = rsum[slot] * __expf(rmax[slot] - nm)
                               + s * __expf(mv - nm);
                    rmax[slot] = nm;
                    rkey[slot] = key > rkey[slot] ? key : rkey[slot];
                } else {
                    float lm = -__builtin_inff(), cap = -__builtin_inff();
#pragma unroll
                    for (int c = 0; c < 8; ++c) {
                        vv[c] = scr[lrow * SCR_ST + lane + 32 * c];
                        int n = n0 + lane + 32 * c;
                        lm = fmaxf(lm, vv[c]);
                        cap = (n == rch[slot]) ? vv[c] : cap;
                    }
#pragma unroll
                    for (int off = 16; off; off >>= 1) {
                        lm  = fmaxf(lm,  __shfl_xor(lm,  off, 32));
                        cap = fmaxf(cap, __shfl_xor(cap, off, 32));
                    }
                    float s = 0.f;
#pragma unroll
                    for (int c = 0; c < 8; ++c) s += __expf(vv[c] - lm);
#pragma unroll
                    for (int off = 16; off; off >>= 1) s += __shfl_xor(s, off, 32);
                    float nm = fmaxf(rmax[slot], lm);
                    rsum[slot] = rsum[slot] * __expf(rmax[slot] - nm)
                               + s * __expf(lm - nm);
                    rmax[slot] = nm;
                    rcap[slot] = fmaxf(rcap[slot], cap);
                }
            }
            __syncthreads();
        }
    }

    if (lane == 0) {
#pragma unroll
        for (int s = 0; s < 4; ++s) {
            int t  = token0 + 16 * (s >> 1) + 2 * wave + (s & 1);
            int pi = t * NSPLIT + blockIdx.y;
            pmax[pi] = rmax[s];
            psum[pi] = rsum[s];
            if (MODE == 0) {
                pbv[pi] = funord((uint32_t)(rkey[s] >> 32));
                pbi[pi] = (int)~(uint32_t)rkey[s];
            } else {
                pcap[pi] = rcap[s];
            }
        }
    }
}

// ---------------- merge N-split partials
__global__ void merge_policy(const float* __restrict__ pm, const float* __restrict__ ps,
                             const float* __restrict__ pbv, const int* __restrict__ pbi,
                             int* __restrict__ chosen, float* __restrict__ lp) {
    int t = blockIdx.x * blockDim.x + threadIdx.x;
    if (t >= TOKENS) return;
    float m = pm[t * NSPLIT], s = ps[t * NSPLIT];
    float bv = pbv[t * NSPLIT]; int bi = pbi[t * NSPLIT];
    for (int k = 1; k < NSPLIT; ++k) {
        float m2 = pm[t * NSPLIT + k], s2 = ps[t * NSPLIT + k];
        float nm = fmaxf(m, m2);
        s = s * __expf(m - nm) + s2 * __expf(m2 - nm);
        m = nm;
        float bv2 = pbv[t * NSPLIT + k];
        if (bv2 > bv) { bv = bv2; bi = pbi[t * NSPLIT + k]; } // first occurrence kept
    }
    chosen[t] = bi;
    lp[t] = bv - (m + __logf(s));
}

__global__ void merge_ref(const float* __restrict__ pm, const float* __restrict__ ps,
                          const float* __restrict__ pc, float* __restrict__ rlp) {
    int t = blockIdx.x * blockDim.x + threadIdx.x;
    if (t >= TOKENS) return;
    float m = pm[t * NSPLIT], s = ps[t * NSPLIT], cap = pc[t * NSPLIT];
    for (int k = 1; k < NSPLIT; ++k) {
        float m2 = pm[t * NSPLIT + k], s2 = ps[t * NSPLIT + k];
        float nm = fmaxf(m, m2);
        s = s * __expf(m - nm) + s2 * __expf(m2 - nm);
        m = nm;
        cap = fmaxf(cap, pc[t * NSPLIT + k]);
    }
    rlp[t] = cap - (m + __logf(s));
}

// ---------------- final scalar GRPO loss
__global__ void grpo_loss_reduce(const float* __restrict__ chosen_lp,
                                 const float* __restrict__ ref_lp,
                                 const float* __restrict__ adv,
                                 const int*   __restrict__ mask,
                                 float* __restrict__ out) {
    __shared__ float ssum[256], smsk[256];
    float acc = 0.f, macc = 0.f;
    for (int t = threadIdx.x; t < TOKENS; t += blockDim.x) {
        float mk = (float)mask[t];
        float lp = chosen_lp[t];
        float rl = ref_lp[t];
        float a  = adv[t / SS];
        float c1 = __expf(lp - lp);                  // exp(lp - stopgrad(lp)) == 1
        float c2 = fminf(fmaxf(c1, 0.8f), 1.2f);
        float ptl = -fminf(c1 * a, c2 * a);
        float d  = rl - lp;
        ptl += 0.1f * (__expf(d) - d - 1.f);          // BETA * KL
        acc  += ptl * mk;
        macc += mk;
    }
    ssum[threadIdx.x] = acc; smsk[threadIdx.x] = macc;
    __syncthreads();
    for (int s = 128; s; s >>= 1) {
        if (threadIdx.x < (unsigned)s) {
            ssum[threadIdx.x] += ssum[threadIdx.x + s];
            smsk[threadIdx.x] += smsk[threadIdx.x + s];
        }
        __syncthreads();
    }
    if (threadIdx.x == 0) out[0] = ssum[0] / fmaxf(smsk[0], 1.f);
}

extern "C" void kernel_launch(void* const* d_in, const int* in_sizes, int n_in,
                              void* d_out, int out_size, void* d_ws, size_t ws_size,
                              hipStream_t stream) {
    (void)in_sizes; (void)n_in; (void)out_size; (void)ws_size;
    const float* x        = (const float*)d_in[0];
    const float* weight   = (const float*)d_in[1];
    const float* bias     = (const float*)d_in[2];
    const float* ref_x    = (const float*)d_in[3];
    const float* ref_w    = (const float*)d_in[4];
    const float* ref_bias = (const float*)d_in[5];
    const float* adv      = (const float*)d_in[6];
    const int*   amask    = (const int*)d_in[7];

    // workspace carve-up
    char* ws = (char*)d_ws;
    const size_t wbytes = (size_t)VDIM * HDIM * 2;      // 131 MB (bf16)
    uint16_t* wbf  = (uint16_t*)ws;  ws += wbytes;
    uint16_t* rwbf = (uint16_t*)ws;  ws += wbytes;
    float* pmax = (float*)ws;  ws += (size_t)TOKENS * NSPLIT * 4;
    float* psum = (float*)ws;  ws += (size_t)TOKENS * NSPLIT * 4;
    float* pbv  = (float*)ws;  ws += (size_t)TOKENS * NSPLIT * 4;
    int*   pbi  = (int*)ws;    ws += (size_t)TOKENS * NSPLIT * 4;
    float* pcap = (float*)ws;  ws += (size_t)TOKENS * NSPLIT * 4;
    int*   chosen    = (int*)ws;   ws += (size_t)TOKENS * 4;
    float* chosen_lp = (float*)ws; ws += (size_t)TOKENS * 4;
    float* ref_lp    = (float*)ws; ws += (size_t)TOKENS * 4;

    // 1) one-time bf16 conversion of both weight matrices (L2-resident after)
    const long wn = (long)VDIM * HDIM;
    cvt_f32_bf16<<<2048, 256, 0, stream>>>(weight, wbf, wn);
    cvt_f32_bf16<<<2048, 256, 0, stream>>>(ref_w, rwbf, wn);

    // 2) fused GEMM + online softmax passes
    const size_t shbytes = (size_t)M_TILE * LDA_H * 2    // 131,584 B  A tile
                         + (size_t)16 * SCR_ST * 4;      //  16,640 B  softmax scratch
    hipFuncSetAttribute((const void*)fused_logits_softmax<0>,
                        hipFuncAttributeMaxDynamicSharedMemorySize, (int)shbytes);
    hipFuncSetAttribute((const void*)fused_logits_softmax<1>,
                        hipFuncAttributeMaxDynamicSharedMemorySize, (int)shbytes);
    dim3 grid(TOKENS / M_TILE, NSPLIT);   // 128 x 5 = 640 workgroups per pass

    fused_logits_softmax<0><<<grid, 256, shbytes, stream>>>(
        x, wbf, bias, nullptr, pmax, psum, pbv, pbi, pcap);
    merge_policy<<<TOKENS / 256, 256, 0, stream>>>(pmax, psum, pbv, pbi,
                                                   chosen, chosen_lp);
    fused_logits_softmax<1><<<grid, 256, shbytes, stream>>>(
        ref_x, rwbf, ref_bias, chosen, pmax, psum, pbv, pbi, pcap);
    merge_ref<<<TOKENS / 256, 256, 0, stream>>>(pmax, psum, pcap, ref_lp);

    // 3) scalar loss
    grpo_loss_reduce<<<1, 256, 0, stream>>>(chosen_lp, ref_lp, adv, amask,
                                            (float*)d_out);
}
